// EGNNEdges_53609781789143
// MI455X (gfx1250) — matrix operand order repaired and checked by reading, not compiled, not run on hardware
//
#include <hip/hip_runtime.h>

// ---------------- sizes ----------------
#define NN   40000
#define NE   640000
#define HD   128
#define DEPTH 4
#define NG   1250
#define TD   32

typedef __bf16 bf16;
typedef __attribute__((ext_vector_type(8)))  __bf16 v8bf;
typedef __attribute__((ext_vector_type(16))) __bf16 v16bf;
typedef __attribute__((ext_vector_type(8)))  float  v8f;
typedef unsigned short u16;
typedef unsigned int   u32;

// ---------------- helpers ----------------
__device__ inline u16 f2bf(float x) {
    u32 u = __float_as_uint(x);
    u += 0x7fffu + ((u >> 16) & 1u);          // round-to-nearest-even
    return (u16)(u >> 16);
}
__device__ inline float fsilu(float x) {
    return x * __builtin_amdgcn_rcpf(1.f + __expf(-x));
}
__device__ inline float fsigm(float x) {
    return __builtin_amdgcn_rcpf(1.f + __expf(-x));
}
__device__ inline v8f wmma_bf16(v16bf a, v16bf b, v8f c) {
    return __builtin_amdgcn_wmma_f32_16x16x32_bf16(false, a, false, b, (short)0, c, false, false);
}
// A fragment (16x32 bf16) from a row-major LDS tile; ISA A-layout:
// lane hi=0: K = kt*32 + {0..7, 16..23}; hi=1: K = kt*32 + {8..15, 24..31}
__device__ inline v16bf load_afrag(const u16* row, int kt, int hi) {
    v8bf lo  = *(const v8bf*)(row + kt * 32 + hi * 8);
    v8bf hi8 = *(const v8bf*)(row + kt * 32 + 16 + hi * 8);
    return __builtin_shufflevector(lo, hi8, 0,1,2,3,4,5,6,7,8,9,10,11,12,13,14,15);
}
// B fragment: packed layout, lane-major 16 bf16 per lane per tile
__device__ inline v16bf load_bfrag(const u16* w, int tile, int lane) {
    return *(const v16bf*)(w + tile * 512 + lane * 16);
}

// ---------------- weight repack: W[K][N] f32 -> packed bf16 B-fragments ----
// B layout (matches A): lane l of tile (kt,nt): n = nt*16 + (l&15),
// element j -> k = kt*32 + (l>>4)*8 + (j&7) + (j>>3)*16
__global__ void repack_w(const float* __restrict__ src, u16* __restrict__ dst,
                         int K, int N) {
    int idx = blockIdx.x * blockDim.x + threadIdx.x;
    if (idx >= K * N) return;
    int NT = N >> 4;
    int t = idx >> 9, e = idx & 511;
    int lane = e >> 4, j = e & 15;
    int kt = t / NT, nt = t % NT;
    int n = nt * 16 + (lane & 15);
    int k = kt * 32 + (lane >> 4) * 8 + (j & 7) + (j >> 3) * 16;
    dst[idx] = f2bf(src[k * N + n]);
}

// ---------------- atom embedding: h = sum_f atom_emb[f, feat] -------------
__global__ void atom_embed(const int* __restrict__ feat, const float* __restrict__ emb,
                           float* __restrict__ h, u16* __restrict__ hbf) {
    int idx = blockIdx.x * blockDim.x + threadIdx.x;
    if (idx >= NN * HD) return;
    int n = idx >> 7, c = idx & 127;
    float v = 0.f;
#pragma unroll
    for (int f = 0; f < 9; ++f) {
        int a = feat[n * 9 + f];
        v += emb[(f * 64 + a) * HD + c];
    }
    h[idx] = v;
    hbf[idx] = f2bf(v);
}

// ---------------- fused edge kernel (the hot loop) ------------------------
// per wave: 16 edges. m_in=[h_src|h_dst|e] staged bf16 in LDS, two WMMA GEMMs,
// soft gate by cross-lane reduction, atomic scatter to msum.
#define EW 8
#define EDGE_SMEM (143872 + EW * 16384)
__global__ __launch_bounds__(256) void edge_kernel(
    const u16* __restrict__ hbf,
    const int* __restrict__ esrc, const int* __restrict__ edst,
    const int* __restrict__ bond_feat, const float* __restrict__ bond_emb,
    const u16* __restrict__ w1pk, const u16* __restrict__ w2pk,
    const float* __restrict__ b1, const float* __restrict__ b2,
    const float* __restrict__ gateW, const float* __restrict__ gateBp,
    float* __restrict__ msum)
{
    extern __shared__ char smem[];
    u16*   sW1   = (u16*)smem;                   // 12*8*512 = 49152 el (96KB)
    u16*   sW2   = (u16*)(smem + 98304);         //  4*8*512 = 16384 el (32KB)
    float* sBond = (float*)(smem + 131072);      // 3*8*128 f32 (12KB)
    float* sGate = (float*)(smem + 143360);      // 128 f32
    char*  wb    = smem + 143872;

    int tid = threadIdx.x;
    for (int i = tid; i < 24576; i += 256) ((u32*)sW1)[i] = ((const u32*)w1pk)[i];
    for (int i = tid; i < 8192;  i += 256) ((u32*)sW2)[i] = ((const u32*)w2pk)[i];
    for (int i = tid; i < 3072;  i += 256) sBond[i] = bond_emb[i];
    if (tid < 128) sGate[tid] = gateW[tid];
    __syncthreads();

    const float gateB = *gateBp;
    int wave = tid >> 5, lane = tid & 31;
    int nl = lane & 15, hi = lane >> 4;
    u16* sMin = (u16*)(wb + wave * 16384);            // 16 x 384 bf16
    u16* sT1  = (u16*)(wb + wave * 16384 + 12288);    // 16 x 128 bf16

    const int ntiles = NE / 16;
    for (int tile = blockIdx.x * EW + wave; tile < ntiles; tile += gridDim.x * EW) {
        int ebase = tile * 16;
        int sIdx = 0, dIdx = 0;
        if (lane < 16) { sIdx = esrc[ebase + lane]; dIdx = edst[ebase + lane]; }

        // ---- stage m_in rows: [src | dst | bond-embed] as bf16 ----
        for (int r = 0; r < 16; ++r) {
            int s = __shfl(sIdx, r);
            int d = __shfl(dIdx, r);
            u16* row = sMin + r * 384;
            ((uint2*)row)[lane]         = ((const uint2*)(hbf + (size_t)s * HD))[lane];
            ((uint2*)(row + 128))[lane] = ((const uint2*)(hbf + (size_t)d * HD))[lane];
            float ax = 0.f, ay = 0.f, az = 0.f, aw = 0.f;
#pragma unroll
            for (int f = 0; f < 3; ++f) {
                int bix = bond_feat[(ebase + r) * 3 + f];
                float4 bv = ((const float4*)(sBond + (f * 8 + bix) * HD))[lane];
                ax += bv.x; ay += bv.y; az += bv.z; aw += bv.w;
            }
            u16* ep = row + 256 + lane * 4;
            ep[0] = f2bf(ax); ep[1] = f2bf(ay); ep[2] = f2bf(az); ep[3] = f2bf(aw);
        }
        asm volatile("s_wait_dscnt 0" ::: "memory");

        // ---- GEMM1: [16,384] @ [384,128] ----
        v8f acc[8];
#pragma unroll
        for (int i = 0; i < 8; ++i) acc[i] = (v8f){0.f,0.f,0.f,0.f,0.f,0.f,0.f,0.f};
        const u16* aRow = sMin + nl * 384;
#pragma unroll
        for (int kt = 0; kt < 12; ++kt) {
            v16bf a = load_afrag(aRow, kt, hi);
#pragma unroll
            for (int nt = 0; nt < 8; ++nt)
                acc[nt] = wmma_bf16(a, load_bfrag(sW1, kt * 8 + nt, lane), acc[nt]);
        }
        // silu(x + b1) -> t1 (bf16, row-major for next A fragment)
#pragma unroll
        for (int nt = 0; nt < 8; ++nt) {
            int n = nt * 16 + nl;
            float bv = b1[n];
#pragma unroll
            for (int r = 0; r < 8; ++r)
                sT1[(r + 8 * hi) * 128 + n] = f2bf(fsilu(acc[nt][r] + bv));
        }
        asm volatile("s_wait_dscnt 0" ::: "memory");

        // ---- GEMM2: [16,128] @ [128,128] ----
        v8f acc2[8];
#pragma unroll
        for (int i = 0; i < 8; ++i) acc2[i] = (v8f){0.f,0.f,0.f,0.f,0.f,0.f,0.f,0.f};
        const u16* a2 = sT1 + nl * 128;
#pragma unroll
        for (int kt = 0; kt < 4; ++kt) {
            v16bf a = load_afrag(a2, kt, hi);
#pragma unroll
            for (int nt = 0; nt < 8; ++nt)
                acc2[nt] = wmma_bf16(a, load_bfrag(sW2, kt * 8 + nt, lane), acc2[nt]);
        }
        // m = silu(x + b2); gate partial dot (m . gateW) per edge row
        float p[8];
#pragma unroll
        for (int r = 0; r < 8; ++r) p[r] = 0.f;
#pragma unroll
        for (int nt = 0; nt < 8; ++nt) {
            int n = nt * 16 + nl;
            float b2v = b2[n], gw = sGate[n];
#pragma unroll
            for (int r = 0; r < 8; ++r) {
                float x = fsilu(acc2[nt][r] + b2v);
                acc2[nt][r] = x;
                p[r] += x * gw;
            }
        }
        // reduce over the 16 lanes that hold one edge row (xor stays in half)
#pragma unroll
        for (int mask = 1; mask < 16; mask <<= 1)
#pragma unroll
            for (int r = 0; r < 8; ++r) p[r] += __shfl_xor(p[r], mask);
        float g8[8];
#pragma unroll
        for (int r = 0; r < 8; ++r) g8[r] = fsigm(p[r] + gateB);

        // ---- gated scatter: msum[dst] += m * gate (L2 fp32 atomics) ----
#pragma unroll
        for (int r = 0; r < 8; ++r) {
            int dd = __shfl(dIdx, r + 8 * hi);
            float* dp = msum + (size_t)dd * HD;
#pragma unroll
            for (int nt = 0; nt < 8; ++nt)
                unsafeAtomicAdd(dp + nt * 16 + nl, acc2[nt][r] * g8[r]);
        }
    }
}

// ---------------- node MLP (update layers + output layer) -----------------
#define NW 8
#define NODE_SMEM (65536 + NW * 16384)
__global__ __launch_bounds__(256) void node_mlp(
    const float* __restrict__ hin, const float* __restrict__ msum,   // msum may be null
    const u16* __restrict__ w1pk, const u16* __restrict__ w2pk,
    const float* __restrict__ b1, const float* __restrict__ b2,
    int residual, int use_relu,
    float* __restrict__ hout, u16* __restrict__ hbf_out)
{
    extern __shared__ char smem[];
    u16* sW1 = (u16*)smem;                 // 16384 el
    u16* sW2 = (u16*)(smem + 32768);       // 16384 el
    char* wb = smem + 65536;

    int tid = threadIdx.x;
    for (int i = tid; i < 8192; i += 256) ((u32*)sW1)[i] = ((const u32*)w1pk)[i];
    for (int i = tid; i < 8192; i += 256) ((u32*)sW2)[i] = ((const u32*)w2pk)[i];
    __syncthreads();

    int wave = tid >> 5, lane = tid & 31;
    int nl = lane & 15, hi = lane >> 4;
    u16*   sIn  = (u16*)(wb + wave * 16384);              // 16x128 bf16
    u16*   sT1  = (u16*)(wb + wave * 16384 + 4096);       // 16x128 bf16
    float* sRes = (float*)(wb + wave * 16384 + 8192);     // 16x128 f32

    const int ntiles = NN / 16;
    for (int tile = blockIdx.x * NW + wave; tile < ntiles; tile += gridDim.x * NW) {
        for (int r = 0; r < 16; ++r) {
            int g = tile * 16 + r;
            float4 hv = ((const float4*)(hin + (size_t)g * HD))[lane];
            float4 u = hv;
            if (msum) {
                float4 mv = ((const float4*)(msum + (size_t)g * HD))[lane];
                u.x += mv.x; u.y += mv.y; u.z += mv.z; u.w += mv.w;
            }
            u16* row = sIn + r * 128 + lane * 4;
            row[0] = f2bf(u.x); row[1] = f2bf(u.y); row[2] = f2bf(u.z); row[3] = f2bf(u.w);
            if (residual) ((float4*)(sRes + r * 128))[lane] = hv;
        }
        asm volatile("s_wait_dscnt 0" ::: "memory");

        v8f acc[8];
#pragma unroll
        for (int i = 0; i < 8; ++i) acc[i] = (v8f){0.f,0.f,0.f,0.f,0.f,0.f,0.f,0.f};
        const u16* aRow = sIn + nl * 128;
#pragma unroll
        for (int kt = 0; kt < 4; ++kt) {
            v16bf a = load_afrag(aRow, kt, hi);
#pragma unroll
            for (int nt = 0; nt < 8; ++nt)
                acc[nt] = wmma_bf16(a, load_bfrag(sW1, kt * 8 + nt, lane), acc[nt]);
        }
#pragma unroll
        for (int nt = 0; nt < 8; ++nt) {
            int n = nt * 16 + nl;
            float bv = b1[n];
#pragma unroll
            for (int r = 0; r < 8; ++r) {
                float x = acc[nt][r] + bv;
                x = use_relu ? fmaxf(x, 0.f) : fsilu(x);
                sT1[(r + 8 * hi) * 128 + n] = f2bf(x);
            }
        }
        asm volatile("s_wait_dscnt 0" ::: "memory");

        v8f acc2[8];
#pragma unroll
        for (int i = 0; i < 8; ++i) acc2[i] = (v8f){0.f,0.f,0.f,0.f,0.f,0.f,0.f,0.f};
        const u16* a2 = sT1 + nl * 128;
#pragma unroll
        for (int kt = 0; kt < 4; ++kt) {
            v16bf a = load_afrag(a2, kt, hi);
#pragma unroll
            for (int nt = 0; nt < 8; ++nt)
                acc2[nt] = wmma_bf16(a, load_bfrag(sW2, kt * 8 + nt, lane), acc2[nt]);
        }
#pragma unroll
        for (int nt = 0; nt < 8; ++nt) {
            int n = nt * 16 + nl;
            float b2v = b2[n];
#pragma unroll
            for (int r = 0; r < 8; ++r) {
                int m = r + 8 * hi;
                int g2 = tile * 16 + m;
                float x = acc2[nt][r] + b2v;
                if (residual) x += sRes[m * 128 + n];
                hout[(size_t)g2 * HD + n] = x;
                if (hbf_out) hbf_out[(size_t)g2 * HD + n] = f2bf(x);
            }
        }
    }
}

// ---------------- readout accumulate: sum / max-key / count ---------------
__global__ void readout_accum(const float* __restrict__ h, const int* __restrict__ n2g,
                              float* __restrict__ sums, u32* __restrict__ mxk,
                              int* __restrict__ cnt) {
    int n = blockIdx.x, c = threadIdx.x;
    int g = n2g[n];
    float v = h[(size_t)n * HD + c];
    unsafeAtomicAdd(&sums[(size_t)g * HD + c], v);
    u32 u = __float_as_uint(v);
    u32 key = (u & 0x80000000u) ? ~u : (u | 0x80000000u);  // order-preserving
    atomicMax(&mxk[(size_t)g * HD + c], key);
    if (c == 0) atomicAdd(&cnt[g], 1);
}
__device__ inline float keydec(u32 k) {
    return (k & 0x80000000u) ? __uint_as_float(k ^ 0x80000000u) : __uint_as_float(~k);
}

// ---------------- readout GEMM: relu([G,384]@W1)@W2 -> [G,32] -------------
#define RW 8
#define RO_SMEM (106496 + RW * 16384)
__global__ __launch_bounds__(256) void readout_kernel(
    const float* __restrict__ sums, const u32* __restrict__ mxk, const int* __restrict__ cnt,
    const u16* __restrict__ w1pk, const u16* __restrict__ w2pk,
    const float* __restrict__ b1, const float* __restrict__ b2,
    float* __restrict__ out)
{
    extern __shared__ char smem[];
    u16* sW1 = (u16*)smem;                 // 49152 el (96KB)
    u16* sW2 = (u16*)(smem + 98304);       // 4096 el (8KB)
    char* wb = smem + 106496;

    int tid = threadIdx.x;
    for (int i = tid; i < 24576; i += 256) ((u32*)sW1)[i] = ((const u32*)w1pk)[i];
    for (int i = tid; i < 2048;  i += 256) ((u32*)sW2)[i] = ((const u32*)w2pk)[i];
    __syncthreads();

    int wave = tid >> 5, lane = tid & 31;
    int nl = lane & 15, hi = lane >> 4;
    u16* sIn = (u16*)(wb + wave * 16384);             // 16x384 bf16
    u16* sT1 = (u16*)(wb + wave * 16384 + 12288);     // 16x128 bf16

    const int ntiles = (NG + 15) / 16;
    for (int tile = blockIdx.x * RW + wave; tile < ntiles; tile += gridDim.x * RW) {
        for (int r = 0; r < 16; ++r) {
            int g = tile * 16 + r;
            u16* row = sIn + r * 384;
            if (g < NG) {
                int cn = cnt[g];
                float inv = 1.f / (float)(cn > 0 ? cn : 1);
                float4 s4 = ((const float4*)(sums + (size_t)g * HD))[lane];
                u16* rs = row + lane * 4;
                rs[0] = f2bf(s4.x); rs[1] = f2bf(s4.y); rs[2] = f2bf(s4.z); rs[3] = f2bf(s4.w);
                u16* rm = row + 128 + lane * 4;
                rm[0] = f2bf(s4.x * inv); rm[1] = f2bf(s4.y * inv);
                rm[2] = f2bf(s4.z * inv); rm[3] = f2bf(s4.w * inv);
                uint4 k4 = ((const uint4*)(mxk + (size_t)g * HD))[lane];
                u16* rx = row + 256 + lane * 4;
                rx[0] = f2bf(cn > 0 ? keydec(k4.x) : 0.f);
                rx[1] = f2bf(cn > 0 ? keydec(k4.y) : 0.f);
                rx[2] = f2bf(cn > 0 ? keydec(k4.z) : 0.f);
                rx[3] = f2bf(cn > 0 ? keydec(k4.w) : 0.f);
            } else {
#pragma unroll
                for (int s = 0; s < 3; ++s) {
                    u16* rz = row + s * 128 + lane * 4;
                    rz[0] = 0; rz[1] = 0; rz[2] = 0; rz[3] = 0;
                }
            }
        }
        asm volatile("s_wait_dscnt 0" ::: "memory");

        v8f acc[8];
#pragma unroll
        for (int i = 0; i < 8; ++i) acc[i] = (v8f){0.f,0.f,0.f,0.f,0.f,0.f,0.f,0.f};
        const u16* aRow = sIn + nl * 384;
#pragma unroll
        for (int kt = 0; kt < 12; ++kt) {
            v16bf a = load_afrag(aRow, kt, hi);
#pragma unroll
            for (int nt = 0; nt < 8; ++nt)
                acc[nt] = wmma_bf16(a, load_bfrag(sW1, kt * 8 + nt, lane), acc[nt]);
        }
#pragma unroll
        for (int nt = 0; nt < 8; ++nt) {
            int n = nt * 16 + nl;
            float bv = b1[n];
#pragma unroll
            for (int r = 0; r < 8; ++r)
                sT1[(r + 8 * hi) * 128 + n] = f2bf(fmaxf(acc[nt][r] + bv, 0.f));
        }
        asm volatile("s_wait_dscnt 0" ::: "memory");

        v8f acc2[2];
#pragma unroll
        for (int i = 0; i < 2; ++i) acc2[i] = (v8f){0.f,0.f,0.f,0.f,0.f,0.f,0.f,0.f};
        const u16* a2 = sT1 + nl * 128;
#pragma unroll
        for (int kt = 0; kt < 4; ++kt) {
            v16bf a = load_afrag(a2, kt, hi);
#pragma unroll
            for (int nt = 0; nt < 2; ++nt)
                acc2[nt] = wmma_bf16(a, load_bfrag(sW2, kt * 2 + nt, lane), acc2[nt]);
        }
#pragma unroll
        for (int nt = 0; nt < 2; ++nt) {
            int n = nt * 16 + nl;
            float b2v = b2[n];
#pragma unroll
            for (int r = 0; r < 8; ++r) {
                int g2 = tile * 16 + r + 8 * hi;
                if (g2 < NG) out[(size_t)g2 * TD + n] = acc2[nt][r] + b2v;
            }
        }
    }
}

// ---------------- host ----------------
extern "C" void kernel_launch(void* const* d_in, const int* in_sizes, int n_in,
                              void* d_out, int out_size, void* d_ws, size_t ws_size,
                              hipStream_t stream) {
    (void)in_sizes; (void)n_in; (void)out_size; (void)ws_size;
    const int*   atom_feat = (const int*)d_in[0];
    const int*   bond_feat = (const int*)d_in[1];
    const int*   esrc      = (const int*)d_in[2];
    const int*   edst      = (const int*)d_in[3];
    const int*   n2g       = (const int*)d_in[4];
    const float* atom_emb  = (const float*)d_in[5];
    const float* bond_emb  = (const float*)d_in[6];
    const float* msg_W1 = (const float*)d_in[7];
    const float* msg_b1 = (const float*)d_in[8];
    const float* msg_W2 = (const float*)d_in[9];
    const float* msg_b2 = (const float*)d_in[10];
    const float* gate_W = (const float*)d_in[11];
    const float* gate_b = (const float*)d_in[12];
    const float* upd_W1 = (const float*)d_in[13];
    const float* upd_b1 = (const float*)d_in[14];
    const float* upd_W2 = (const float*)d_in[15];
    const float* upd_b2 = (const float*)d_in[16];
    const float* out_W1 = (const float*)d_in[17];
    const float* out_b1 = (const float*)d_in[18];
    const float* out_W2 = (const float*)d_in[19];
    const float* out_b2 = (const float*)d_in[20];
    const float* ro_W1  = (const float*)d_in[21];
    const float* ro_b1  = (const float*)d_in[22];
    const float* ro_W2  = (const float*)d_in[23];
    const float* ro_b2  = (const float*)d_in[24];
    float* out = (float*)d_out;

    char* ws = (char*)d_ws;
    size_t off = 0;
    auto alloc = [&](size_t bytes) -> char* {
        char* p = ws + off;
        off += (bytes + 255) & ~(size_t)255;
        return p;
    };
    float* h    = (float*)alloc((size_t)NN * HD * 4);
    u16*   hbf  = (u16*)  alloc((size_t)NN * HD * 2);
    float* msum = (float*)alloc((size_t)NN * HD * 4);
    u16* msgW1pk = (u16*)alloc((size_t)DEPTH * 384 * 128 * 2);
    u16* msgW2pk = (u16*)alloc((size_t)DEPTH * 128 * 128 * 2);
    u16* updW1pk = (u16*)alloc((size_t)DEPTH * 128 * 128 * 2);
    u16* updW2pk = (u16*)alloc((size_t)DEPTH * 128 * 128 * 2);
    u16* outW1pk = (u16*)alloc(128 * 128 * 2);
    u16* outW2pk = (u16*)alloc(128 * 128 * 2);
    u16* roW1pk  = (u16*)alloc(384 * 128 * 2);
    u16* roW2pk  = (u16*)alloc(128 * 32 * 2);
    float* sums  = (float*)alloc((size_t)NG * HD * 4);
    u32*   mxk   = (u32*)  alloc((size_t)NG * HD * 4);
    int*   cnt   = (int*)  alloc((size_t)NG * 4);

    (void)hipFuncSetAttribute((const void*)edge_kernel,
        hipFuncAttributeMaxDynamicSharedMemorySize, EDGE_SMEM);
    (void)hipFuncSetAttribute((const void*)node_mlp,
        hipFuncAttributeMaxDynamicSharedMemorySize, NODE_SMEM);
    (void)hipFuncSetAttribute((const void*)readout_kernel,
        hipFuncAttributeMaxDynamicSharedMemorySize, RO_SMEM);

    auto rp = [&](const float* s, u16* dpk, int K, int Nn) {
        int tot = K * Nn;
        repack_w<<<(tot + 255) / 256, 256, 0, stream>>>(s, dpk, K, Nn);
    };
    for (int l = 0; l < DEPTH; ++l) {
        rp(msg_W1 + (size_t)l * 384 * 128, msgW1pk + (size_t)l * 384 * 128, 384, 128);
        rp(msg_W2 + (size_t)l * 128 * 128, msgW2pk + (size_t)l * 128 * 128, 128, 128);
        rp(upd_W1 + (size_t)l * 128 * 128, updW1pk + (size_t)l * 128 * 128, 128, 128);
        rp(upd_W2 + (size_t)l * 128 * 128, updW2pk + (size_t)l * 128 * 128, 128, 128);
    }
    rp(out_W1, outW1pk, 128, 128);
    rp(out_W2, outW2pk, 128, 128);
    rp(ro_W1, roW1pk, 384, 128);
    rp(ro_W2, roW2pk, 128, 32);

    atom_embed<<<(NN * HD + 255) / 256, 256, 0, stream>>>(atom_feat, atom_emb, h, hbf);

    for (int l = 0; l < DEPTH; ++l) {
        (void)hipMemsetAsync(msum, 0, (size_t)NN * HD * 4, stream);
        edge_kernel<<<640, 256, EDGE_SMEM, stream>>>(
            hbf, esrc, edst, bond_feat, bond_emb,
            msgW1pk + (size_t)l * 384 * 128, msgW2pk + (size_t)l * 128 * 128,
            msg_b1 + l * 128, msg_b2 + l * 128,
            gate_W + l * 128, gate_b + l, msum);
        node_mlp<<<313, 256, NODE_SMEM, stream>>>(
            h, msum,
            updW1pk + (size_t)l * 128 * 128, updW2pk + (size_t)l * 128 * 128,
            upd_b1 + l * 128, upd_b2 + l * 128,
            /*residual=*/1, /*relu=*/0, h, hbf);
    }
    // node-wise output MLP (no residual), in-place
    node_mlp<<<313, 256, NODE_SMEM, stream>>>(
        h, nullptr, outW1pk, outW2pk, out_b1, out_b2,
        /*residual=*/0, /*relu=*/0, h, nullptr);

    (void)hipMemsetAsync(sums, 0, (size_t)NG * HD * 4, stream);
    (void)hipMemsetAsync(mxk, 0, (size_t)NG * HD * 4, stream);   // key 0 < key(-inf)
    (void)hipMemsetAsync(cnt, 0, (size_t)NG * 4, stream);
    readout_accum<<<NN, 128, 0, stream>>>(h, n2g, sums, mxk, cnt);
    readout_kernel<<<10, 256, RO_SMEM, stream>>>(sums, mxk, cnt,
        roW1pk, roW2pk, ro_b1, ro_b2, out);
}
